// MobileViTV2Block_84619445666717
// MI455X (gfx1250) — compile-verified, matrix operand order checked by reading, not run on hardware
//
#include <hip/hip_runtime.h>
#include <hip/hip_bf16.h>

typedef __attribute__((ext_vector_type(16))) __bf16 v16bf;
typedef __attribute__((ext_vector_type(8)))  __bf16 v8bf;
typedef __attribute__((ext_vector_type(8)))  float  v8f;

// Problem constants
// B=8, C=256, T=16, H=W=32, D=384, FF=768, OUT=256, tokens M = 8*4*4096 = 131072

__device__ __forceinline__ float sigmoidf_(float x) { return 1.f / (1.f + expf(-x)); }

// Load a 16x32 bf16 WMMA A/B fragment. Lane L holds row (L&15); two contiguous
// 16-byte K-chunks at K offsets 8*(L>>4) and 16+8*(L>>4)  (CDNA5 ISA 7.12.2).
__device__ __forceinline__ v16bf ldfrag(const __bf16* p) {
  v8bf lo = *(const v8bf*)(p);
  v8bf hi = *(const v8bf*)(p + 16);
  return __builtin_shufflevector(lo, hi, 0,1,2,3,4,5,6,7,8,9,10,11,12,13,14,15);
}

// ---------------------------------------------------------------- depthwise conv + SiLU
// writes bf16 activations in token-major layout y[token][c] so pw1 GEMM is dense
__global__ __launch_bounds__(256) void dwconv_silu_kernel(
    const float* __restrict__ x, const float* __restrict__ w,
    const float* __restrict__ bias, __bf16* __restrict__ y) {
  int idx = blockIdx.x * 256 + threadIdx.x;            // 33,554,432 total
  int wv = idx & 31;
  int h  = (idx >> 5) & 31;
  int t  = (idx >> 10) & 15;
  int c  = (idx >> 14) & 255;
  int b  = idx >> 22;
  const float* xc = x + ((size_t)b * 256 + c) * 16384;
  const float* wc = w + c * 27;
  float acc = 0.f;
  #pragma unroll
  for (int dt = -1; dt <= 1; ++dt) {
    int tt = t + dt; if (tt < 0 || tt > 15) continue;
    #pragma unroll
    for (int dh = -1; dh <= 1; ++dh) {
      int hh = h + dh; if (hh < 0 || hh > 31) continue;
      #pragma unroll
      for (int dw = -1; dw <= 1; ++dw) {
        int ww = wv + dw; if (ww < 0 || ww > 31) continue;
        acc += xc[tt * 1024 + hh * 32 + ww] * wc[(dt + 1) * 9 + (dh + 1) * 3 + (dw + 1)];
      }
    }
  }
  acc += bias[c];
  float s = acc * sigmoidf_(acc);
  int token = (b << 14) | (((h & 1) * 2 + (wv & 1)) << 12) | (t << 8) | ((h >> 1) << 4) | (wv >> 1);
  y[(size_t)token * 256 + c] = (__bf16)s;
}

// ---------------------------------------------------------------- bf16 WMMA GEMM
// C[M,N] = A[M,K] * Bt[N,K]^T (+bias) with fused epilogues.
// Block: 8 waves as 4(M) x 2(N); wave tile 32x64 (2x4 WMMA tiles) -> block tile 128x128.
// Per K-step: 6 fragment loads (12 x b128) feed 8 v_wmma (compute:load = 1.33).
constexpr int EPI_ZSTORE = 0;  // outF[row*N+col]  = val
constexpr int EPI_ZADD   = 1;  // outF[row*N+col] += val (residual)
constexpr int EPI_BF16   = 2;  // outB[row*N+col]  = bf16(val)
constexpr int EPI_SILU   = 3;  // outB[row*N+col]  = bf16(silu(val))
constexpr int EPI_PW2    = 4;  // scatter to (B,OUT,T,H,W) f32

template<int EPI>
__global__ __launch_bounds__(256) void gemm_bf16_kernel(
    const __bf16* __restrict__ A, const __bf16* __restrict__ Bt,
    const float* __restrict__ bias, float* __restrict__ outF,
    __bf16* __restrict__ outB, int K, int N) {
  const int lane  = threadIdx.x & 31;
  const int wave  = threadIdx.x >> 5;
  const int mBase = blockIdx.x * 128 + (wave & 3) * 32;   // 4 M-wave positions
  const int nBase = blockIdx.y * 128 + (wave >> 2) * 64;  // 2 N-wave positions
  const int r  = lane & 15;
  const int hh = lane >> 4;

  const __bf16* aP0 = A  + (size_t)(mBase + r) * K + 8 * hh;
  const __bf16* aP1 = aP0 + (size_t)16 * K;
  const __bf16* bP  = Bt + (size_t)(nBase + r) * K + 8 * hh;

  v8f acc[2][4] = {};
  for (int kc = 0; kc < K; kc += 32) {
    v16bf a0 = ldfrag(aP0 + kc);
    v16bf a1 = ldfrag(aP1 + kc);
    v16bf b0 = ldfrag(bP + kc);
    v16bf b1 = ldfrag(bP + (size_t)16 * K + kc);
    v16bf b2 = ldfrag(bP + (size_t)32 * K + kc);
    v16bf b3 = ldfrag(bP + (size_t)48 * K + kc);
    acc[0][0] = __builtin_amdgcn_wmma_f32_16x16x32_bf16(false, a0, false, b0, (short)0, acc[0][0], false, false);
    acc[0][1] = __builtin_amdgcn_wmma_f32_16x16x32_bf16(false, a0, false, b1, (short)0, acc[0][1], false, false);
    acc[0][2] = __builtin_amdgcn_wmma_f32_16x16x32_bf16(false, a0, false, b2, (short)0, acc[0][2], false, false);
    acc[0][3] = __builtin_amdgcn_wmma_f32_16x16x32_bf16(false, a0, false, b3, (short)0, acc[0][3], false, false);
    acc[1][0] = __builtin_amdgcn_wmma_f32_16x16x32_bf16(false, a1, false, b0, (short)0, acc[1][0], false, false);
    acc[1][1] = __builtin_amdgcn_wmma_f32_16x16x32_bf16(false, a1, false, b1, (short)0, acc[1][1], false, false);
    acc[1][2] = __builtin_amdgcn_wmma_f32_16x16x32_bf16(false, a1, false, b2, (short)0, acc[1][2], false, false);
    acc[1][3] = __builtin_amdgcn_wmma_f32_16x16x32_bf16(false, a1, false, b3, (short)0, acc[1][3], false, false);
  }
  #pragma unroll
  for (int i = 0; i < 2; ++i) {
    #pragma unroll
    for (int j = 0; j < 4; ++j) {
      int col = nBase + j * 16 + r;
      float bv = bias[col];
      #pragma unroll
      for (int v = 0; v < 8; ++v) {
        int row = mBase + i * 16 + hh * 8 + v;  // C layout: lane<16 -> M=v, lane>=16 -> M=8+v
        float val = acc[i][j][v] + bv;
        size_t o = (size_t)row * N + col;
        if constexpr (EPI == EPI_ZSTORE) {
          outF[o] = val;
        } else if constexpr (EPI == EPI_ZADD) {
          outF[o] += val;
        } else if constexpr (EPI == EPI_BF16) {
          outB[o] = (__bf16)val;
        } else if constexpr (EPI == EPI_SILU) {
          outB[o] = (__bf16)(val * sigmoidf_(val));
        } else {  // EPI_PW2: token -> (b, col, t, h, w)
          int b  = row >> 14;
          int p  = (row >> 12) & 3;
          int n  = row & 4095;
          int t  = n >> 8;
          int hp = (n >> 4) & 15;
          int wp = n & 15;
          int h  = hp * 2 + (p >> 1);
          int w  = wp * 2 + (p & 1);
          outF[((((size_t)b * 256 + col) * 16 + t) * 32 + h) * 32 + w] = val;
        }
      }
    }
  }
}

// ---------------------------------------------------------------- LayerNorm (wave per row)
__global__ __launch_bounds__(256) void ln_bf16_kernel(
    const float* __restrict__ z, const float* __restrict__ g,
    const float* __restrict__ bta, __bf16* __restrict__ out) {
  int lane = threadIdx.x & 31;
  size_t row = (size_t)blockIdx.x * 8 + (threadIdx.x >> 5);
  const float* zr = z + row * 384;
  float vals[12], s = 0.f, ss = 0.f;
  #pragma unroll
  for (int j = 0; j < 12; ++j) { float v = zr[lane + 32 * j]; vals[j] = v; s += v; ss += v * v; }
  for (int m = 16; m >= 1; m >>= 1) { s += __shfl_xor(s, m, 32); ss += __shfl_xor(ss, m, 32); }
  float mean = s * (1.f / 384.f);
  float rs = rsqrtf(ss * (1.f / 384.f) - mean * mean + 1e-5f);
  #pragma unroll
  for (int j = 0; j < 12; ++j) {
    int d = lane + 32 * j;
    out[row * 384 + d] = (__bf16)((vals[j] - mean) * rs * g[d] + bta[d]);
  }
}

__global__ __launch_bounds__(256) void cvt_f32_bf16_kernel(
    const float* __restrict__ z, __bf16* __restrict__ out) {
  size_t i = (size_t)blockIdx.x * 256 + threadIdx.x;
  out[i] = (__bf16)z[i];
}

// ---------------------------------------------------------------- q GEMV (wave per row)
__global__ __launch_bounds__(256) void qgemv_kernel(
    const __bf16* __restrict__ h, const float* __restrict__ wq,
    const float* __restrict__ qb, float* __restrict__ q) {
  int lane = threadIdx.x & 31;
  size_t row = (size_t)blockIdx.x * 8 + (threadIdx.x >> 5);
  float s = 0.f;
  #pragma unroll
  for (int j = 0; j < 12; ++j) {
    int d = lane + 32 * j;
    s += (float)h[row * 384 + d] * wq[(size_t)d * 769];
  }
  for (int m = 16; m >= 1; m >>= 1) s += __shfl_xor(s, m, 32);
  if (lane == 0) q[row] = s + qb[0];
}

// ---------------------------------------------------------------- softmax(q) + context vector
// one block per (b,p) group of 4096 tokens; 384 threads, thread d accumulates cv[d]
__global__ __launch_bounds__(384) void softcv_kernel(
    const float* __restrict__ q, const __bf16* __restrict__ kv, float* __restrict__ cv) {
  __shared__ float wts[4096];
  __shared__ float red[384];
  __shared__ float s_mx, s_inv;
  int tid = threadIdx.x, gp = blockIdx.x;
  const float* qg = q + (size_t)gp * 4096;
  float lm = -1e30f;
  for (int n = tid; n < 4096; n += 384) lm = fmaxf(lm, qg[n]);
  red[tid] = lm; __syncthreads();
  if (tid == 0) { float m = -1e30f; for (int i = 0; i < 384; ++i) m = fmaxf(m, red[i]); s_mx = m; }
  __syncthreads();
  float mx = s_mx, ls = 0.f;
  for (int n = tid; n < 4096; n += 384) { float e = expf(qg[n] - mx); wts[n] = e; ls += e; }
  red[tid] = ls; __syncthreads();
  if (tid == 0) { float t = 0.f; for (int i = 0; i < 384; ++i) t += red[i]; s_inv = 1.f / t; }
  __syncthreads();
  float acc = 0.f;
  const __bf16* kg = kv + (size_t)gp * 4096 * 768 + tid;  // k part: cols [0,384)
  for (int n = 0; n < 4096; ++n) acc += wts[n] * (float)kg[(size_t)n * 768];
  cv[gp * 384 + tid] = acc * s_inv;
}

// a = relu(v) * cv[group]
__global__ __launch_bounds__(256) void relu_cv_kernel(
    const __bf16* __restrict__ kv, const float* __restrict__ cv, __bf16* __restrict__ a) {
  size_t idx = (size_t)blockIdx.x * 256 + threadIdx.x;
  int row = (int)(idx / 384), d = (int)(idx % 384);
  float v = (float)kv[(size_t)row * 768 + 384 + d];  // v part: cols [384,768)
  a[idx] = (__bf16)(fmaxf(v, 0.f) * cv[(row >> 12) * 384 + d]);
}

// ---------------------------------------------------------------- weight repack: dst[n*K+k] = src[k*rowStride + colOff + n]
__global__ __launch_bounds__(256) void repack_kernel(
    const float* __restrict__ src, __bf16* __restrict__ dst,
    int N, int K, int rowStride, int colOff) {
  int idx = blockIdx.x * 256 + threadIdx.x;
  if (idx >= N * K) return;
  int n = idx / K, k = idx % K;
  dst[idx] = (__bf16)src[(size_t)k * rowStride + colOff + n];
}

// ---------------------------------------------------------------- orchestration
extern "C" void kernel_launch(void* const* d_in, const int* in_sizes, int n_in,
                              void* d_out, int out_size, void* d_ws, size_t ws_size,
                              hipStream_t stream) {
  (void)in_sizes; (void)n_in; (void)out_size; (void)ws_size;
  const float* x     = (const float*)d_in[0];
  const float* dw_w  = (const float*)d_in[1];
  const float* dw_b  = (const float*)d_in[2];
  const float* pw1_w = (const float*)d_in[3];
  const float* pw1_b = (const float*)d_in[4];
  const float* ln1_g = (const float*)d_in[5];
  const float* ln1_b = (const float*)d_in[6];
  const float* qkv_w = (const float*)d_in[7];
  const float* qkv_b = (const float*)d_in[8];
  const float* wo_w  = (const float*)d_in[9];
  const float* wo_b  = (const float*)d_in[10];
  const float* ln2_g = (const float*)d_in[11];
  const float* ln2_b = (const float*)d_in[12];
  const float* ff1_w = (const float*)d_in[13];
  const float* ff1_b = (const float*)d_in[14];
  const float* ff2_w = (const float*)d_in[15];
  const float* ff2_b = (const float*)d_in[16];
  const float* pw2_w = (const float*)d_in[17];
  const float* pw2_b = (const float*)d_in[18];

  const size_t M = 131072;  // tokens
  char* ws = (char*)d_ws;
  size_t off = 0;
  auto alloc = [&](size_t bytes) { void* p = ws + off; off += (bytes + 255) & ~(size_t)255; return p; };
  float*  z   = (float*) alloc(M * 384 * 4);   // residual stream (f32)
  __bf16* hbf = (__bf16*)alloc(M * 384 * 2);   // LN output h, later reused as 'a'
  __bf16* kvg = (__bf16*)alloc(M * 768 * 2);   // conv-out ybf, then kv, then ff hidden g
  float*  qv  = (float*) alloc(M * 4);         // q scores
  float*  cv  = (float*) alloc(32 * 384 * 4);  // context vectors per (b,p)
  __bf16* wp  = (__bf16*)alloc((size_t)2400000 * 2);  // packed weights

  __bf16* pw1t = wp;
  __bf16* pw2t = pw1t + 384 * 256;
  __bf16* kvt[2], *wot[2], *f1t[2], *f2t[2];
  {
    __bf16* c = pw2t + 256 * 384;
    for (int i = 0; i < 2; ++i) {
      kvt[i] = c; c += 768 * 384;
      wot[i] = c; c += 384 * 384;
      f1t[i] = c; c += 768 * 384;
      f2t[i] = c; c += 384 * 768;
    }
  }

  auto rp = [&](const float* s, __bf16* d, int N, int K, int rs, int co) {
    int tot = N * K;
    repack_kernel<<<(tot + 255) / 256, 256, 0, stream>>>(s, d, N, K, rs, co);
  };
  rp(pw1_w, pw1t, 384, 256, 384, 0);
  rp(pw2_w, pw2t, 256, 384, 256, 0);
  for (int i = 0; i < 2; ++i) {
    rp(qkv_w + (size_t)i * 384 * 769, kvt[i], 768, 384, 769, 1);  // cols 1..768 = [k|v]
    rp(wo_w  + (size_t)i * 384 * 384, wot[i], 384, 384, 384, 0);
    rp(ff1_w + (size_t)i * 384 * 768, f1t[i], 768, 384, 768, 0);
    rp(ff2_w + (size_t)i * 768 * 384, f2t[i], 384, 768, 384, 0);
  }

  __bf16* ybf = kvg;  // conv output aliases kv buffer (dead before kv GEMM)
  dwconv_silu_kernel<<<131072, 256, 0, stream>>>(x, dw_w, dw_b, ybf);

  // pw1: z = ybf @ pw1 + b   (K=256, N=384)
  gemm_bf16_kernel<EPI_ZSTORE><<<dim3(1024, 3), 256, 0, stream>>>(ybf, pw1t, pw1_b, z, nullptr, 256, 384);

  for (int i = 0; i < 2; ++i) {
    ln_bf16_kernel<<<16384, 256, 0, stream>>>(z, ln1_g + i * 384, ln1_b + i * 384, hbf);
    qgemv_kernel<<<16384, 256, 0, stream>>>(hbf, qkv_w + (size_t)i * 384 * 769, qkv_b + i * 769, qv);
    // [k|v] = h @ qkv_w[:,1:] + b   (K=384, N=768)
    gemm_bf16_kernel<EPI_BF16><<<dim3(1024, 6), 256, 0, stream>>>(hbf, kvt[i], qkv_b + i * 769 + 1, nullptr, kvg, 384, 768);
    softcv_kernel<<<32, 384, 0, stream>>>(qv, kvg, cv);
    relu_cv_kernel<<<196608, 256, 0, stream>>>(kvg, cv, hbf);  // hbf := a
    // z += a @ wo + b   (K=384, N=384)
    gemm_bf16_kernel<EPI_ZADD><<<dim3(1024, 3), 256, 0, stream>>>(hbf, wot[i], wo_b + i * 384, z, nullptr, 384, 384);
    ln_bf16_kernel<<<16384, 256, 0, stream>>>(z, ln2_g + i * 384, ln2_b + i * 384, hbf);
    // g = silu(h @ ff1 + b)   (K=384, N=768)
    gemm_bf16_kernel<EPI_SILU><<<dim3(1024, 6), 256, 0, stream>>>(hbf, f1t[i], ff1_b + i * 768, nullptr, kvg, 384, 768);
    // z += g @ ff2 + b   (K=768, N=384)
    gemm_bf16_kernel<EPI_ZADD><<<dim3(1024, 3), 256, 0, stream>>>(kvg, f2t[i], ff2_b + i * 384, z, nullptr, 768, 384);
  }

  // out = z @ pw2 + b, scattered back to (B, OUT, T, H, W)   (K=384, N=256)
  cvt_f32_bf16_kernel<<<196608, 256, 0, stream>>>(z, hbf);
  gemm_bf16_kernel<EPI_PW2><<<dim3(1024, 2), 256, 0, stream>>>(hbf, pw2t, pw2_b, (float*)d_out, nullptr, 384, 256);
}
